// DGCNN_3layers_rotate_90228672954732
// MI455X (gfx1250) — matrix operand order repaired and checked by reading, not compiled, not run on hardware
//
#include <hip/hip_runtime.h>
#include <math.h>

// ---------------- problem constants ----------------
constexpr int NB   = 8;          // batches
constexpr int NP   = 2048;       // points
constexpr int KNN  = 20;         // neighbors
constexpr int BN   = NB * NP;            // 16384
constexpr int BNK  = BN * KNN;           // 327680
constexpr int OUTC = 448;                // 64+128+256
constexpr float INV_P  = 1.0f / (float)BNK;
constexpr float BN_EPS = 1e-5f;

typedef __bf16  v16bf __attribute__((ext_vector_type(16)));
typedef __bf16  v8bf  __attribute__((ext_vector_type(8)));
typedef float   v8f   __attribute__((ext_vector_type(8)));

#define WMMA_BF16(A, Bv, Cv) \
  __builtin_amdgcn_wmma_f32_16x16x32_bf16(false, (A), false, (Bv), (short)0, (Cv), false, false)

// load two contiguous 8-element bf16 runs (16B each -> global_load_b128) and
// concatenate into the 16-element WMMA fragment
__device__ __forceinline__ v16bf ld16(const __bf16* p0, const __bf16* p1) {
  v8bf a = *reinterpret_cast<const v8bf*>(p0);
  v8bf b = *reinterpret_cast<const v8bf*>(p1);
  return __builtin_shufflevector(a, b, 0, 1, 2, 3, 4, 5, 6, 7, 8, 9, 10, 11, 12, 13, 14, 15);
}

// =======================================================================
// 0) zero the BN-stat accumulators (896 floats: s1/q1 s2/q2 s3/q3)
// =======================================================================
__global__ void k_zero_stats(float* stats) {
  int t = blockIdx.x * 256 + threadIdx.x;
  if (t < 896) stats[t] = 0.0f;
}

// =======================================================================
// 1) convert conv2_w / conv3_w to bf16 hi/lo split
// =======================================================================
__global__ void k_cvt_weights(const float* w2, const float* w3,
                              __bf16* w2h, __bf16* w2l, __bf16* w3h, __bf16* w3l) {
  int e = blockIdx.x * 256 + threadIdx.x;
  if (e < 128 * 128) {
    float f = w2[e]; __bf16 h = (__bf16)f;
    w2h[e] = h; w2l[e] = (__bf16)(f - (float)h);
  }
  int e3 = e - 128 * 128;
  if (e3 >= 0 && e3 < 256 * 256) {
    float f = w3[e3]; __bf16 h = (__bf16)f;
    w3h[e3] = h; w3l[e3] = (__bf16)(f - (float)h);
  }
}

// =======================================================================
// 2) brute-force kNN on 3-D coords.  Batch's points staged in LDS (24KB);
//    one thread per query point; 20 exact selection passes with
//    top_k-compatible tie-breaking (desc value, asc index).
// =======================================================================
__global__ __launch_bounds__(256) void k_knn3(const float* __restrict__ x, int* __restrict__ idx1) {
  __shared__ float sx[NP * 3];
  int b = blockIdx.x >> 3;
  int g = blockIdx.x & 7;
  const float* xb = x + (size_t)b * NP * 3;
  for (int t = threadIdx.x; t < NP * 3; t += 256) sx[t] = xb[t];
  __syncthreads();
  int n = g * 256 + threadIdx.x;
  float px = sx[n * 3 + 0], py = sx[n * 3 + 1], pz = sx[n * 3 + 2];
  float xxn = px * px + py * py + pz * pz;
  float lastV = 3.4e38f; int lastM = -1;
  int* out = idx1 + ((size_t)b * NP + n) * KNN;
  for (int p = 0; p < KNN; ++p) {
    float bestV = -3.4e38f; int bestM = 0;
    for (int m = 0; m < NP; ++m) {
      float qx = sx[m * 3 + 0], qy = sx[m * 3 + 1], qz = sx[m * 3 + 2];
      float dot = px * qx + py * qy + pz * qz;
      float v = 2.0f * dot - xxn - (qx * qx + qy * qy + qz * qz);
      bool ok = (v < lastV) || (v == lastV && m > lastM);
      if (ok && v > bestV) { bestV = v; bestM = m; }   // '>' keeps lowest index on ties
    }
    out[p] = bestM;
    lastV = bestV; lastM = bestM;
  }
}

// =======================================================================
// 3) geometric feature build: R1 (Rodrigues to z), R2 (mean -> +x),
//    mirror; writes f1 (B,N,K,9) and R1,R2 into d_out.
// =======================================================================
__global__ __launch_bounds__(256) void k_feat1(const float* __restrict__ x,
                                               const float* __restrict__ nrm,
                                               const int* __restrict__ idx1,
                                               float* __restrict__ f1,
                                               float* __restrict__ outR1,
                                               float* __restrict__ outR2) {
  int bn = blockIdx.x * 256 + threadIdx.x;     // 16384
  int b = bn >> 11, n = bn & (NP - 1);
  const float* xb = x + (size_t)b * NP * 3;
  float px = xb[n * 3 + 0], py = xb[n * 3 + 1], pz = xb[n * 3 + 2];
  float nx = nrm[(size_t)bn * 3 + 0], ny = nrm[(size_t)bn * 3 + 1], nz = nrm[(size_t)bn * 3 + 2];
  // v = n x z = (ny, -nx, 0);  R1 = I + K + K^2 * s
  float vx = ny, vy = -nx, c = nz;
  float s = 1.0f / fmaxf(1.0f + c, 1e-8f);
  float R[3][3];
  R[0][0] = 1.0f - vy * vy * s; R[0][1] = vx * vy * s;        R[0][2] = vy;
  R[1][0] = vx * vy * s;        R[1][1] = 1.0f - vx * vx * s; R[1][2] = -vx;
  R[2][0] = -vy;                R[2][1] = vx;                 R[2][2] = 1.0f - (vx * vx + vy * vy) * s;

  const int* id = idx1 + (size_t)bn * KNN;
  float mx = 0.0f, my = 0.0f;
  for (int k = 0; k < KNN; ++k) {
    int nb = id[k];
    float dx = xb[nb * 3 + 0] - px, dy = xb[nb * 3 + 1] - py, dz = xb[nb * 3 + 2] - pz;
    mx += R[0][0] * dx + R[0][1] * dy + R[0][2] * dz;
    my += R[1][0] * dx + R[1][1] * dy + R[1][2] * dz;
  }
  mx *= (1.0f / KNN); my *= (1.0f / KNN);
  float rl = sqrtf(mx * mx + my * my);
  float ct = 1.0f, st = 0.0f;
  if (rl > 0.0f) { ct = mx / rl; st = my / rl; }  // cos/sin(atan2(my,mx))

  for (int k = 0; k < KNN; ++k) {
    int nb = id[k];
    float dx = xb[nb * 3 + 0] - px, dy = xb[nb * 3 + 1] - py, dz = xb[nb * 3 + 2] - pz;
    float r0 = R[0][0] * dx + R[0][1] * dy + R[0][2] * dz;
    float r1 = R[1][0] * dx + R[1][1] * dy + R[1][2] * dz;
    float r2 = R[2][0] * dx + R[2][1] * dy + R[2][2] * dz;
    float f0 = ct * r0 + st * r1;       // R2 * rot
    float f1v = -st * r0 + ct * r1;
    float* fp = f1 + ((size_t)bn * KNN + k) * 9;
    fp[0] = f0;  fp[1] = f1v;  fp[2] = r2;      // rot2
    fp[3] = f0;  fp[4] = -f1v; fp[5] = r2;      // xz-mirror
    fp[6] = px;  fp[7] = py;   fp[8] = pz;      // x_rep
  }
  float* r1o = outR1 + (size_t)bn * 9;
  for (int i = 0; i < 3; ++i)
    for (int j = 0; j < 3; ++j) r1o[i * 3 + j] = R[i][j];
  float* r2o = outR2 + (size_t)bn * 9;
  r2o[0] = ct; r2o[1] = st;  r2o[2] = 0.0f;
  r2o[3] = -st; r2o[4] = ct; r2o[5] = 0.0f;
  r2o[6] = 0.0f; r2o[7] = 0.0f; r2o[8] = 1.0f;
}

// =======================================================================
// 4) conv1 (64x9, K too small for WMMA): stats pass then BN+LReLU+max pass
// =======================================================================
__global__ __launch_bounds__(256) void k_conv1_stats(const float* __restrict__ f1,
                                                     const float* __restrict__ w1,
                                                     float* __restrict__ gs, float* __restrict__ gq) {
  int bn = blockIdx.x * 256 + threadIdx.x;
  int lane = threadIdx.x & 31;
  const float* f = f1 + (size_t)bn * (KNN * 9);
  for (int o = 0; o < 64; ++o) {
    float w[9];
#pragma unroll
    for (int c = 0; c < 9; ++c) w[c] = w1[o * 9 + c];
    float s = 0.0f, q = 0.0f;
    for (int k = 0; k < KNN; ++k) {
      float h = 0.0f;
#pragma unroll
      for (int c = 0; c < 9; ++c) h += w[c] * f[k * 9 + c];
      s += h; q += h * h;
    }
    for (int d = 1; d < 32; d <<= 1) { s += __shfl_xor(s, d, 32); q += __shfl_xor(q, d, 32); }
    if (lane == 0) { atomicAdd(&gs[o], s); atomicAdd(&gq[o], q); }
  }
}

__global__ __launch_bounds__(256) void k_conv1_out(const float* __restrict__ f1,
                                                   const float* __restrict__ w1,
                                                   const float* __restrict__ gs, const float* __restrict__ gq,
                                                   const float* __restrict__ gg, const float* __restrict__ gb,
                                                   float* __restrict__ x1ws, float* __restrict__ out) {
  int bn = blockIdx.x * 256 + threadIdx.x;
  const float* f = f1 + (size_t)bn * (KNN * 9);
  for (int o = 0; o < 64; ++o) {
    float w[9];
#pragma unroll
    for (int c = 0; c < 9; ++c) w[c] = w1[o * 9 + c];
    float mean = gs[o] * INV_P;
    float var  = gq[o] * INV_P - mean * mean;
    float sc = gg[o] * rsqrtf(var + BN_EPS);
    float sh = gb[o] - mean * sc;
    float best = -3.4e38f;
    for (int k = 0; k < KNN; ++k) {
      float h = 0.0f;
#pragma unroll
      for (int c = 0; c < 9; ++c) h += w[c] * f[k * 9 + c];
      float y = h * sc + sh;
      y = (y > 0.0f) ? y : 0.2f * y;
      best = fmaxf(best, y);
    }
    x1ws[(size_t)bn * 64 + o] = best;
    out[(size_t)bn * OUTC + o] = best;
  }
}

// =======================================================================
// 5) graph-conv layers 2/3 (WMMA bf16-split GEMM fused with gather).
//    One block per (b,n); WAVES waves cover COUT in 16-row tiles; the 20
//    K-positions are padded to two 16-column WMMA tiles (padding zeroed).
// =======================================================================
template<int C, int COUT, int WAVES, bool STATS>
__global__ __launch_bounds__(WAVES * 32)
void k_gconv(const float* __restrict__ xin, const int* __restrict__ idx,
             const __bf16* __restrict__ Wh, const __bf16* __restrict__ Wl,
             float* __restrict__ gsum, float* __restrict__ gsq,
             const float* __restrict__ gg, const float* __restrict__ gb,
             float* __restrict__ xout, float* __restrict__ out, int chanOff) {
  constexpr int CIN = 2 * C;
  __shared__ __align__(16) float sctr[C];
  __shared__ __align__(16) float snbr[KNN * C];
  __shared__ float sSum[COUT], sSq[COUT];

  int bn = blockIdx.x;
  int b = bn >> 11;
  int tid = threadIdx.x;
  if (STATS) for (int t = tid; t < COUT; t += WAVES * 32) { sSum[t] = 0.0f; sSq[t] = 0.0f; }
  const int* id = idx + (size_t)bn * KNN;
  for (int t = tid; t < (KNN + 1) * C; t += WAVES * 32) {
    if (t < C) sctr[t] = xin[(size_t)bn * C + t];
    else {
      int j = (t - C) / C, c = (t - C) % C;
      int nb = id[j];
      snbr[j * C + c] = xin[((size_t)(b << 11) + nb) * C + c];
    }
  }
  __syncthreads();

  int wave = tid >> 5, lane = tid & 31;
  int coutBase = wave * 16;
  int nloc = lane & 15;
  bool hiHalf = lane >= 16;
  int offA = hiHalf ? 16 : 0;    // A-fragment: runs at kb+offA and kb+offA+8
  int offB = hiHalf ? 16 : 0;    // B-fragment: one run of 16 at kb+offB

  v8f acc0 = {0.f, 0.f, 0.f, 0.f, 0.f, 0.f, 0.f, 0.f};
  v8f acc1 = {0.f, 0.f, 0.f, 0.f, 0.f, 0.f, 0.f, 0.f};

  for (int kb = 0; kb < CIN; kb += 32) {
    // A-tile: 16 Cout rows x 32 channels, two contiguous b128 loads per frag
    int m = coutBase + nloc;
    const __bf16* wr = Wh + (size_t)m * CIN + kb + offA;
    const __bf16* wl = Wl + (size_t)m * CIN + kb + offA;
    v16bf ah = ld16(wr, wr + 8);
    v16bf al = ld16(wl, wl + 8);

    // B-tiles: 32 channels x 16 positions, built from LDS-staged gathers.
    // The 16-channel run never straddles the diff/center boundary.
#pragma unroll
    for (int t = 0; t < 2; ++t) {
      int pos = t * 16 + nloc;
      float msk = (pos < KNN) ? 1.0f : 0.0f;
      int posc = (pos < KNN) ? pos : 0;
      int cc0 = kb + offB;
      float fv[16];
      if (cc0 < C) {
#pragma unroll
        for (int j = 0; j < 16; ++j) fv[j] = snbr[posc * C + cc0 + j] - sctr[cc0 + j];
      } else {
#pragma unroll
        for (int j = 0; j < 16; ++j) fv[j] = sctr[cc0 - C + j];
      }
      v16bf bh, bl;
#pragma unroll
      for (int j = 0; j < 16; ++j) {
        float v = fv[j] * msk;
        __bf16 h = (__bf16)v;
        bh[j] = h; bl[j] = (__bf16)(v - (float)h);
      }
      v8f& acc = t ? acc1 : acc0;
      acc = WMMA_BF16(ah, bh, acc);
      acc = WMMA_BF16(ah, bl, acc);
      acc = WMMA_BF16(al, bh, acc);
    }
  }

  bool v1 = nloc < (KNN - 16);   // tile-1 positions 16..19 valid
  if (STATS) {
#pragma unroll
    for (int i = 0; i < 8; ++i) {
      float h0 = acc0[i], h1 = acc1[i];
      float si = h0 + (v1 ? h1 : 0.0f);
      float qi = h0 * h0 + (v1 ? h1 * h1 : 0.0f);
      for (int d = 1; d < 16; d <<= 1) { si += __shfl_xor(si, d, 32); qi += __shfl_xor(qi, d, 32); }
      if (nloc == 0) {
        int mm = coutBase + i + (hiHalf ? 8 : 0);
        atomicAdd(&sSum[mm], si); atomicAdd(&sSq[mm], qi);
      }
    }
    __syncthreads();
    for (int t = tid; t < COUT; t += WAVES * 32) {
      atomicAdd(&gsum[t], sSum[t]); atomicAdd(&gsq[t], sSq[t]);
    }
  } else {
#pragma unroll
    for (int i = 0; i < 8; ++i) {
      int mm = coutBase + i + (hiHalf ? 8 : 0);
      float mean = gsum[mm] * INV_P;
      float var  = gsq[mm] * INV_P - mean * mean;
      float sc = gg[mm] * rsqrtf(var + BN_EPS);
      float sh = gb[mm] - mean * sc;
      float y0 = acc0[i] * sc + sh; y0 = (y0 > 0.0f) ? y0 : 0.2f * y0;
      float best = y0;
      if (v1) { float y1 = acc1[i] * sc + sh; y1 = (y1 > 0.0f) ? y1 : 0.2f * y1; best = fmaxf(best, y1); }
      for (int d = 1; d < 16; d <<= 1) best = fmaxf(best, __shfl_xor(best, d, 32));
      if (nloc == 0) {
        if (xout) xout[(size_t)bn * COUT + mm] = best;
        out[(size_t)bn * OUTC + chanOff + mm] = best;
      }
    }
  }
}

// =======================================================================
// 6) x2 -> bf16 hi/lo split + squared norms (for feature-space kNN)
// =======================================================================
__global__ void k_cvt_x2(const float* __restrict__ x2, __bf16* __restrict__ xh, __bf16* __restrict__ xl) {
  int e = blockIdx.x * 256 + threadIdx.x;
  if (e < BN * 128) {
    float f = x2[e]; __bf16 h = (__bf16)f;
    xh[e] = h; xl[e] = (__bf16)(f - (float)h);
  }
}
__global__ void k_xx2(const float* __restrict__ x2, float* __restrict__ xx) {
  int bn = blockIdx.x * 256 + threadIdx.x;
  const float* p = x2 + (size_t)bn * 128;
  float s = 0.0f;
  for (int c = 0; c < 128; ++c) s += p[c] * p[c];
  xx[bn] = s;
}

// =======================================================================
// 7) per-batch 2048x2048x128 distance GEMM via WMMA bf16-split.
//    One wave per 16x16 tile (12 wmma each); all fragments loaded as b128.
// =======================================================================
__global__ __launch_bounds__(256) void k_dist(const __bf16* __restrict__ xh, const __bf16* __restrict__ xl,
                                              const float* __restrict__ xx, float* __restrict__ dist, int b) {
  int wave = threadIdx.x >> 5, lane = threadIdx.x & 31;
  int tile = blockIdx.x * 8 + wave;          // 16384 tiles
  int ti = tile >> 7, tj = tile & 127;
  int R = ti * 16, Cb = tj * 16;
  const __bf16* Xh = xh + (size_t)b * NP * 128;
  const __bf16* Xl = xl + (size_t)b * NP * 128;
  const float* xxb = xx + (size_t)b * NP;
  int nloc = lane & 15; bool hiHalf = lane >= 16;
  int offA = hiHalf ? 16 : 0;
  int offB = hiHalf ? 16 : 0;
  int m0 = R + nloc, n0 = Cb + nloc;

  v8f acc = {0.f, 0.f, 0.f, 0.f, 0.f, 0.f, 0.f, 0.f};
  for (int kb = 0; kb < 128; kb += 32) {
    const __bf16* pah = Xh + (size_t)m0 * 128 + kb + offA;
    const __bf16* pal = Xl + (size_t)m0 * 128 + kb + offA;
    const __bf16* pbh = Xh + (size_t)n0 * 128 + kb + offB;
    const __bf16* pbl = Xl + (size_t)n0 * 128 + kb + offB;
    v16bf ah = ld16(pah, pah + 8);
    v16bf al = ld16(pal, pal + 8);
    v16bf bh = ld16(pbh, pbh + 8);
    v16bf bl = ld16(pbl, pbl + 8);
    acc = WMMA_BF16(ah, bh, acc);
    acc = WMMA_BF16(ah, bl, acc);
    acc = WMMA_BF16(al, bh, acc);
  }
  float xc = xxb[Cb + nloc];
#pragma unroll
  for (int i = 0; i < 8; ++i) {
    int m = R + i + (hiHalf ? 8 : 0);
    dist[(size_t)m * NP + Cb + nloc] = 2.0f * acc[i] - xxb[m] - xc;
  }
}

// =======================================================================
// 8) wave-per-row exact top-20 over the L2-resident distance row
// =======================================================================
__global__ __launch_bounds__(256) void k_top20(const float* __restrict__ dist, int* __restrict__ idx2, int b) {
  int wave = threadIdx.x >> 5, lane = threadIdx.x & 31;
  int row = blockIdx.x * 8 + wave;
  const float* dr = dist + (size_t)row * NP;
  float lastV = 3.4e38f; int lastM = -1;
  int* out = idx2 + ((size_t)b * NP + row) * KNN;
  for (int p = 0; p < KNN; ++p) {
    float bestV = -3.4e38f; int bestM = -1;
    for (int m = lane; m < NP; m += 32) {
      float v = dr[m];
      bool ok = (v < lastV) || (v == lastV && m > lastM);
      if (ok && v > bestV) { bestV = v; bestM = m; }
    }
    for (int d = 1; d < 32; d <<= 1) {
      float ov = __shfl_xor(bestV, d, 32);
      int   om = __shfl_xor(bestM, d, 32);
      if (ov > bestV || (ov == bestV && (unsigned)om < (unsigned)bestM)) { bestV = ov; bestM = om; }
    }
    if (lane == 0) out[p] = bestM;
    lastV = bestV; lastM = bestM;
  }
}

// =======================================================================
// host launcher
// =======================================================================
extern "C" void kernel_launch(void* const* d_in, const int* in_sizes, int n_in,
                              void* d_out, int out_size, void* d_ws, size_t ws_size,
                              hipStream_t stream) {
  (void)in_sizes; (void)n_in; (void)out_size; (void)ws_size;
  const float* x       = (const float*)d_in[0];
  const float* normals = (const float*)d_in[1];
  const float* conv1_w = (const float*)d_in[2];
  const float* bn1_g   = (const float*)d_in[3];
  const float* bn1_b   = (const float*)d_in[4];
  const float* conv2_w = (const float*)d_in[5];
  const float* bn2_g   = (const float*)d_in[6];
  const float* bn2_b   = (const float*)d_in[7];
  const float* conv3_w = (const float*)d_in[8];
  const float* bn3_g   = (const float*)d_in[9];
  const float* bn3_b   = (const float*)d_in[10];
  float* out = (float*)d_out;

  // workspace layout (256B aligned)
  char* ws = (char*)d_ws;
  size_t o = 0;
  auto take = [&](size_t bytes) { size_t r = o; o += (bytes + 255) & ~(size_t)255; return r; };
  int*    idx1  = (int*)   (ws + take((size_t)BNK * 4));
  int*    idx2  = (int*)   (ws + take((size_t)BNK * 4));
  float*  x1ws  = (float*) (ws + take((size_t)BN * 64 * 4));
  float*  x2ws  = (float*) (ws + take((size_t)BN * 128 * 4));
  float*  f1    = (float*) (ws + take((size_t)BNK * 9 * 4));
  float*  stats = (float*) (ws + take(896 * 4));
  __bf16* w2h   = (__bf16*)(ws + take(128 * 128 * 2));
  __bf16* w2l   = (__bf16*)(ws + take(128 * 128 * 2));
  __bf16* w3h   = (__bf16*)(ws + take(256 * 256 * 2));
  __bf16* w3l   = (__bf16*)(ws + take(256 * 256 * 2));
  __bf16* x2h   = (__bf16*)(ws + take((size_t)BN * 128 * 2));
  __bf16* x2l   = (__bf16*)(ws + take((size_t)BN * 128 * 2));
  float*  xx2   = (float*) (ws + take((size_t)BN * 4));
  float*  dist  = (float*) (ws + take((size_t)NP * NP * 4));

  float* gs1 = stats;       float* gq1 = stats + 64;
  float* gs2 = stats + 128; float* gq2 = stats + 256;
  float* gs3 = stats + 384; float* gq3 = stats + 640;

  float* outR1 = out + (size_t)BN * OUTC;
  float* outR2 = outR1 + (size_t)BN * 9;

  k_zero_stats<<<4, 256, 0, stream>>>(stats);
  k_cvt_weights<<<(128 * 128 + 256 * 256 + 255) / 256, 256, 0, stream>>>(conv2_w, conv3_w, w2h, w2l, w3h, w3l);

  // layer 1: coordinate kNN -> geometric features -> conv1 + BN + LReLU + max
  k_knn3<<<NB * 8, 256, 0, stream>>>(x, idx1);
  k_feat1<<<BN / 256, 256, 0, stream>>>(x, normals, idx1, f1, outR1, outR2);
  k_conv1_stats<<<BN / 256, 256, 0, stream>>>(f1, conv1_w, gs1, gq1);
  k_conv1_out<<<BN / 256, 256, 0, stream>>>(f1, conv1_w, gs1, gq1, bn1_g, bn1_b, x1ws, out);

  // layer 2: graph features on x1 (reuse coordinate kNN), 128x128 WMMA conv
  k_gconv<64, 128, 8, true ><<<BN, 256, 0, stream>>>(x1ws, idx1, w2h, w2l, gs2, gq2, nullptr, nullptr, nullptr, nullptr, 0);
  k_gconv<64, 128, 8, false><<<BN, 256, 0, stream>>>(x1ws, idx1, w2h, w2l, gs2, gq2, bn2_g, bn2_b, x2ws, out, 64);

  // feature-space kNN on x2: per-batch N^2 distance GEMM (WMMA) + exact top-20
  k_cvt_x2<<<(BN * 128 + 255) / 256, 256, 0, stream>>>(x2ws, x2h, x2l);
  k_xx2<<<BN / 256, 256, 0, stream>>>(x2ws, xx2);
  for (int b = 0; b < NB; ++b) {
    k_dist <<<(NP / 16) * (NP / 16) / 8, 256, 0, stream>>>(x2h, x2l, xx2, dist, b);
    k_top20<<<NP / 8, 256, 0, stream>>>(dist, idx2, b);
  }

  // layer 3: graph features on x2 with fresh kNN, 256x256 WMMA conv
  k_gconv<128, 256, 16, true ><<<BN, 512, 0, stream>>>(x2ws, idx2, w3h, w3l, gs3, gq3, nullptr, nullptr, nullptr, nullptr, 0);
  k_gconv<128, 256, 16, false><<<BN, 512, 0, stream>>>(x2ws, idx2, w3h, w3l, gs3, gq3, bn3_g, bn3_b, nullptr, out, 192);
}
